// ACRoPEAttention_2516850836111
// MI455X (gfx1250) — compile-verified, matrix operand order checked by reading, not run on hardware
//
#include <hip/hip_runtime.h>
#include <hip/hip_fp16.h>

// ---------------------------------------------------------------------------
// CDNA5 (gfx1250) attention block:
//   qkv = x @ Wqkv^T            (WMMA f16 GEMM, f32 accum, async LDS staging)
//   q,k <- axial RoPE, split heads, q pre-scaled by hd^-0.5
//   o   = softmax(q k^T) v      (flash attention; S computed TRANSPOSED so
//                                softmax reductions are in-lane; WMMA f16)
//   out = o @ Wproj^T + b       (WMMA f16 GEMM, f32 out)
// ---------------------------------------------------------------------------

typedef __attribute__((ext_vector_type(16))) _Float16 v16h;
typedef __attribute__((ext_vector_type(8)))  _Float16 v8h;
typedef __attribute__((ext_vector_type(4)))  _Float16 v4h;
typedef __attribute__((ext_vector_type(8)))  float    v8f;

static __device__ __forceinline__ v8f wmma_f16(v16h a, v16h b, v8f c) {
  // D = A(16x32) * B(32x16) + C, f32 accumulate
  return __builtin_amdgcn_wmma_f32_16x16x32_f16(false, a, false, b, (short)0, c,
                                                false, false);
}

// Load a 16x32 A-fragment (or 32x16 B-fragment, by symmetry) from a row-major
// tile in LDS/global.  Lane l holds row r = l&15; lane-half selects k-offset 8.
// v16h element j: j<8 -> k = koff+j ; j>=8 -> k = 16+koff+(j-8).
static __device__ __forceinline__ v16h frag_from(const _Float16* base, int ld,
                                                 int lane, int kbase) {
  const int r    = lane & 15;
  const int koff = ((lane >> 4) << 3) + kbase;
  const _Float16* p = base + r * ld + koff;
  v8h lo = *(const v8h*)p;
  v8h hi = *(const v8h*)(p + 16);
  return __builtin_shufflevector(lo, hi, 0, 1, 2, 3, 4, 5, 6, 7, 8, 9, 10, 11,
                                 12, 13, 14, 15);
}

// Async global->LDS 128-bit copy (ASYNCcnt-tracked, ISA §10 GV mode).
// VDST vgpr = per-lane LDS byte offset (low 32 bits of generic LDS pointer),
// VADDR = 64-bit global address.
static __device__ __forceinline__ void async_b128(_Float16* lds,
                                                  const _Float16* gsrc) {
  asm volatile("global_load_async_to_lds_b128 %0, %1, off"
               :
               : "v"((unsigned)(unsigned long long)lds), "v"(gsrc)
               : "memory");
}
static __device__ __forceinline__ void wait_async0() {
  asm volatile("s_wait_asynccnt 0" ::: "memory");
}

// ---------------------------------------------------------------------------
// fp32 -> f16 conversion (vectorized by 4)
// ---------------------------------------------------------------------------
__global__ __launch_bounds__(256) void cvt_f32_f16_kernel(
    const float* __restrict__ in, _Float16* __restrict__ out, int n4) {
  int i = blockIdx.x * blockDim.x + threadIdx.x;
  if (i < n4) {
    float4 f = ((const float4*)in)[i];
    v4h h = {(_Float16)f.x, (_Float16)f.y, (_Float16)f.z, (_Float16)f.w};
    ((v4h*)out)[i] = h;
  }
}

// ---------------------------------------------------------------------------
// WMMA GEMM:  C[M,Nout] = A[M,K] * W[Nout,K]^T  (+bias)
// Block tile 64x128, 8 waves (wave32): wave (wm, wn) computes 16x64.
// Tiles staged with async global->LDS copies.
// REMAP_A: A is attention output laid out [B*nH][N][hd] but logically [M][K]
//          with m=(b,n), k=(h,d).
// ---------------------------------------------------------------------------
template <bool REMAP_A, bool HAS_BIAS, bool OUT_HALF>
__global__ __launch_bounds__(256) void gemm_wmma_kernel(
    const _Float16* __restrict__ A, const _Float16* __restrict__ W,
    const float* __restrict__ bias, void* __restrict__ Cout, int M, int K,
    int Nout) {
  __shared__ __align__(16) _Float16 As[64 * 32];
  __shared__ __align__(16) _Float16 Bs[128 * 32];
  const int m0   = blockIdx.x * 64;
  const int n0   = blockIdx.y * 128;
  const int w    = threadIdx.x >> 5;
  const int lane = threadIdx.x & 31;
  const int wm   = w & 3;          // 16-row strip
  const int wn   = (w >> 2) * 64;  // 64-col strip

  v8f acc[4];
#pragma unroll
  for (int t = 0; t < 4; ++t)
#pragma unroll
    for (int r = 0; r < 8; ++r) acc[t][r] = 0.0f;

  for (int k0 = 0; k0 < K; k0 += 32) {
    {  // stage A tile 64x32 (256 thr * one async b128 each)
      int idx = threadIdx.x * 8;
      int r = idx >> 5, c = idx & 31;
      int gm = m0 + r, gk = k0 + c;
      const _Float16* src;
      if (REMAP_A) {
        int b = gm >> 11, n = gm & 2047, h = gk >> 6, d = gk & 63;
        src = A + ((((size_t)b * 16 + h) * 2048 + n) * 64 + d);
      } else {
        src = A + (size_t)gm * K + gk;
      }
      async_b128(As + idx, src);
    }
#pragma unroll
    for (int i = 0; i < 2; ++i) {  // stage B tile 128x32
      int idx = threadIdx.x * 16 + i * 8;
      int r = idx >> 5, c = idx & 31;
      async_b128(Bs + idx, W + (size_t)(n0 + r) * K + k0 + c);
    }
    wait_async0();
    __syncthreads();

    v16h af = frag_from(As + wm * 16 * 32, 32, lane, 0);
#pragma unroll
    for (int t = 0; t < 4; ++t) {
      v16h bf = frag_from(Bs + (wn + t * 16) * 32, 32, lane, 0);
      acc[t] = wmma_f16(af, bf, acc[t]);
    }
    __syncthreads();
  }

  // Epilogue: C layout lane l -> n = l&15, vgpr r -> m = r + 8*(l>>4)
  const int half = lane >> 4;
  const int nl   = lane & 15;
#pragma unroll
  for (int t = 0; t < 4; ++t)
#pragma unroll
    for (int r = 0; r < 8; ++r) {
      int gm = m0 + wm * 16 + r + 8 * half;
      int gn = n0 + wn + t * 16 + nl;
      float v = acc[t][r];
      if (HAS_BIAS) v += bias[gn];
      if (OUT_HALF)
        ((_Float16*)Cout)[(size_t)gm * Nout + gn] = (_Float16)v;
      else
        ((float*)Cout)[(size_t)gm * Nout + gn] = v;
    }
}

// ---------------------------------------------------------------------------
// RoPE + head split.  qkv: [B][N][3][nH][hd] (f16).  Outputs q/k/v as
// [B*nH][N][hd] f16, q pre-scaled by hd^-0.5, q/k rotated (d,h,w segments).
// ---------------------------------------------------------------------------
static __device__ __forceinline__ void rotate_seg(float* xs, int len,
                                                  float pos) {
  const int half = len >> 1;
#pragma unroll
  for (int i = 0; i < 10; ++i) {  // half==10 for hd=64
    if (i >= half) break;
    float x0 = xs[2 * i], x1 = xs[2 * i + 1];
    // tiled (not interleaved) sin/cos: index j%half for j = 2i, 2i+1
    float f0 = pos * powf(10000.0f, -(float)((2 * i) % half) / (float)half);
    float f1 = pos * powf(10000.0f, -(float)((2 * i + 1) % half) / (float)half);
    xs[2 * i]     = x0 * cosf(f0) - x1 * sinf(f0);
    xs[2 * i + 1] = x1 * cosf(f1) + x0 * sinf(f1);
  }
}

__global__ __launch_bounds__(256) void rope_split_kernel(
    const _Float16* __restrict__ qkv, _Float16* __restrict__ qh,
    _Float16* __restrict__ kh, _Float16* __restrict__ vh,
    const int* __restrict__ Tp, const int* __restrict__ Hp,
    const int* __restrict__ Wp) {
  (void)Tp;
  const int N = 2048, nH = 16, hd = 64;
  int idx = blockIdx.x * blockDim.x + threadIdx.x;
  if (idx >= 2 * nH * N) return;
  int n = idx % N;
  int h = (idx / N) % nH;
  int b = idx / (N * nH);

  int Hq = *Hp, Wq = *Wp;
  int HW = Hq * Wq;
  float dpos = (float)(n / HW);
  int rem = n % HW;
  float hpos = (float)(rem / Wq) * (16.0f / (float)Hq);
  float wpos = (float)(rem % Wq) * (16.0f / (float)Wq);
  const int dd = 2 * ((hd / 3) / 2);  // 20

  size_t src = ((size_t)(b * N + n) * 3) * (nH * hd) + (size_t)h * hd;
  size_t dst = ((size_t)(b * nH + h) * N + n) * hd;

  float xq[64], xk[64];
#pragma unroll
  for (int d = 0; d < hd; ++d) {
    xq[d] = (float)qkv[src + d];
    xk[d] = (float)qkv[src + 1024 + d];
    vh[dst + d] = qkv[src + 2048 + d];  // v passthrough
  }
  float pos3[3] = {dpos, hpos, wpos};
  for (int s = 0; s < 3; ++s) {
    rotate_seg(xq + s * dd, dd, pos3[s]);
    rotate_seg(xk + s * dd, dd, pos3[s]);
  }
  const float scale = 0.125f;  // 64^-0.5 folded into q
#pragma unroll
  for (int d = 0; d < hd; ++d) {
    qh[dst + d] = (_Float16)(xq[d] * scale);
    kh[dst + d] = (_Float16)xk[d];
  }
}

// ---------------------------------------------------------------------------
// Flash attention, one (b,h): N=2048, hd=64.  4 waves / block, 64 q-rows per
// block (16 per wave), k/v blocks of 64 streamed through LDS.
//
// S is computed TRANSPOSED (S^T = K * Q^T): in the C-tile layout a query is a
// lane column, so the softmax max/sum are in-lane reductions over 32 registers
// plus one xor-16 shuffle; only the O-accumulator rescale needs 8 lane
// broadcasts.  P^T staging to LDS is 4x ds_store_b128 per lane.
// ---------------------------------------------------------------------------
__global__ __launch_bounds__(128) void flash_attn_kernel(
    const _Float16* __restrict__ Q, const _Float16* __restrict__ K,
    const _Float16* __restrict__ V, _Float16* __restrict__ O) {
  const int N = 2048, hd = 64;
  __shared__ __align__(16) _Float16 Ks[64 * 64];     // [key][d] row-major
  __shared__ __align__(16) _Float16 Vts[64 * 64];    // transposed: [d][key]
  __shared__ __align__(16) _Float16 Ps[4][16 * 64];  // per-wave P tile [q][key]

  const int w    = threadIdx.x >> 5;
  const int lane = threadIdx.x & 31;
  const int half = lane >> 4;
  const int nl   = lane & 15;
  const int bh   = blockIdx.y;
  const size_t base = (size_t)bh * N * hd;
  const int m0 = blockIdx.x * 64 + w * 16;

  // Q as B-fragments (dim-chunk x queries): lane holds query row nl.
  v16h qb0, qb1;
  {
    const _Float16* qp = Q + base + (size_t)(m0 + nl) * hd + (half << 3);
    v8h a0 = *(const v8h*)qp;
    v8h a1 = *(const v8h*)(qp + 16);
    v8h a2 = *(const v8h*)(qp + 32);
    v8h a3 = *(const v8h*)(qp + 48);
    qb0 = __builtin_shufflevector(a0, a1, 0, 1, 2, 3, 4, 5, 6, 7, 8, 9, 10, 11,
                                  12, 13, 14, 15);
    qb1 = __builtin_shufflevector(a2, a3, 0, 1, 2, 3, 4, 5, 6, 7, 8, 9, 10, 11,
                                  12, 13, 14, 15);
  }

  v8f acc[4];
#pragma unroll
  for (int t = 0; t < 4; ++t)
#pragma unroll
    for (int r = 0; r < 8; ++r) acc[t][r] = 0.0f;
  float mi = -3.0e38f;  // running max for this lane's query column
  float li = 0.0f;      // running sum

  for (int kb = 0; kb < N; kb += 64) {
    // ---- stage K block via async global->LDS; V transposed via registers ---
    const _Float16* Kg = K + base + (size_t)kb * hd;
#pragma unroll
    for (int it = 0; it < 4; ++it) {
      int linear = it * 1024 + threadIdx.x * 8;
      async_b128(Ks + linear, Kg + linear);
    }
    {
      int key = threadIdx.x >> 1;
      int d0  = (threadIdx.x & 1) * 32;
      const _Float16* vg = V + base + (size_t)(kb + key) * hd + d0;
      v8h vr[4];
#pragma unroll
      for (int j4 = 0; j4 < 4; ++j4) vr[j4] = ((const v8h*)vg)[j4];
#pragma unroll
      for (int j = 0; j < 32; ++j)
        Vts[(d0 + j) * 64 + key] = vr[j >> 3][j & 7];
    }
    if (kb + 64 < N) __builtin_prefetch(Kg + 64 * hd, 0, 1);
    wait_async0();
    __syncthreads();

    // ---- S^T = K * Q^T : rows = keys, cols(lanes) = queries ----
    v8f st[4];
#pragma unroll
    for (int t = 0; t < 4; ++t) {
#pragma unroll
      for (int r = 0; r < 8; ++r) st[t][r] = 0.0f;
      st[t] = wmma_f16(frag_from(Ks + t * 16 * 64, 64, lane, 0), qb0, st[t]);
      st[t] = wmma_f16(frag_from(Ks + t * 16 * 64, 64, lane, 32), qb1, st[t]);
    }

    // ---- online softmax: in-lane over 32 keys + one xor-16 combine ----
    float mt = st[0][0];
#pragma unroll
    for (int t = 0; t < 4; ++t)
#pragma unroll
      for (int r = 0; r < 8; ++r) mt = fmaxf(mt, st[t][r]);
    mt = fmaxf(mt, __shfl_xor(mt, 16, 32));
    float newm = fmaxf(mi, mt);
    float corrq = __expf(mi - newm);
    mi = newm;
    float ps = 0.0f;
#pragma unroll
    for (int t = 0; t < 4; ++t)
#pragma unroll
      for (int r = 0; r < 8; ++r) {
        float p = __expf(st[t][r] - newm);
        st[t][r] = p;
        ps += p;
      }
    ps += __shfl_xor(ps, 16, 32);
    li = li * corrq + ps;

    // rescale O accumulator: its rows are queries -> broadcast corr per row
    float corrb[8];
#pragma unroll
    for (int r = 0; r < 8; ++r) corrb[r] = __shfl(corrq, r + 8 * half, 32);
#pragma unroll
    for (int t = 0; t < 4; ++t)
#pragma unroll
      for (int r = 0; r < 8; ++r) acc[t][r] *= corrb[r];

    // ---- stage P (row-major [query][key]) : 4x b128 per lane ----
    _Float16* pw = &Ps[w][0];
#pragma unroll
    for (int t = 0; t < 4; ++t) {
      v8h ph;
#pragma unroll
      for (int r = 0; r < 8; ++r) ph[r] = (_Float16)st[t][r];
      *(v8h*)(pw + nl * 64 + t * 16 + 8 * half) = ph;
    }
    asm volatile("s_wait_dscnt 0" ::: "memory");  // wave-local LDS RAW fence

    v16h pf0 = frag_from(pw, 64, lane, 0);
    v16h pf1 = frag_from(pw, 64, lane, 32);
#pragma unroll
    for (int t = 0; t < 4; ++t) {
      acc[t] = wmma_f16(pf0, frag_from(Vts + t * 16 * 64, 64, lane, 0), acc[t]);
      acc[t] = wmma_f16(pf1, frag_from(Vts + t * 16 * 64, 64, lane, 32), acc[t]);
    }
    __syncthreads();
  }

  // ---- epilogue: O /= l (broadcast per row), write [BH][N][hd] f16 ----
  float lib[8];
#pragma unroll
  for (int r = 0; r < 8; ++r) lib[r] = __shfl(li, r + 8 * half, 32);
#pragma unroll
  for (int t = 0; t < 4; ++t)
#pragma unroll
    for (int r = 0; r < 8; ++r) {
      int gm = m0 + r + 8 * half;
      int d  = t * 16 + nl;
      O[base + (size_t)gm * hd + d] = (_Float16)(acc[t][r] / lib[r]);
    }
}

// ---------------------------------------------------------------------------
// Host-side orchestration
// ---------------------------------------------------------------------------
extern "C" void kernel_launch(void* const* d_in, const int* in_sizes, int n_in,
                              void* d_out, int out_size, void* d_ws,
                              size_t ws_size, hipStream_t stream) {
  (void)in_sizes; (void)n_in; (void)out_size; (void)ws_size;
  const float* x     = (const float*)d_in[0];  // [2,2048,1024]
  const float* Wqkv  = (const float*)d_in[1];  // [3072,1024]
  const float* Wproj = (const float*)d_in[2];  // [1024,1024]
  const float* bproj = (const float*)d_in[3];  // [1024]
  const int* Tp = (const int*)d_in[4];
  const int* Hp = (const int*)d_in[5];
  const int* Wp = (const int*)d_in[6];

  const int B = 2, N = 2048, C = 1024, nH = 16, hd = 64;
  const int M = B * N;    // 4096
  const int BH = B * nH;  // 32

  // workspace carve-up (all f16, all 16B aligned; ~76 MB total)
  _Float16* xh     = (_Float16*)d_ws;
  _Float16* wqkvh  = xh + (size_t)M * C;
  _Float16* wprojh = wqkvh + (size_t)3 * C * C;
  _Float16* qkvh   = wprojh + (size_t)C * C;
  _Float16* qh     = qkvh + (size_t)M * 3 * C;
  _Float16* kh     = qh + (size_t)BH * N * hd;
  _Float16* vh     = kh + (size_t)BH * N * hd;
  _Float16* oh     = vh + (size_t)BH * N * hd;

  // 1) fp32 -> f16 conversions
  {
    int n4 = (M * C) / 4;
    cvt_f32_f16_kernel<<<(n4 + 255) / 256, 256, 0, stream>>>(x, xh, n4);
    n4 = (3 * C * C) / 4;
    cvt_f32_f16_kernel<<<(n4 + 255) / 256, 256, 0, stream>>>(Wqkv, wqkvh, n4);
    n4 = (C * C) / 4;
    cvt_f32_f16_kernel<<<(n4 + 255) / 256, 256, 0, stream>>>(Wproj, wprojh, n4);
  }

  // 2) QKV projection: qkv[4096,3072] = xh * Wqkv^T
  gemm_wmma_kernel<false, false, true>
      <<<dim3(M / 64, (3 * C) / 128), 256, 0, stream>>>(
          xh, wqkvh, nullptr, (void*)qkvh, M, C, 3 * C);

  // 3) RoPE + split heads (q pre-scaled)
  rope_split_kernel<<<(BH * N + 255) / 256, 256, 0, stream>>>(
      qkvh, qh, kh, vh, Tp, Hp, Wp);

  // 4) Flash attention per (b,h)
  flash_attn_kernel<<<dim3(N / 64, BH), 128, 0, stream>>>(qh, kh, vh, oh);

  // 5) Output projection + bias: d_out[4096,1024] f32
  gemm_wmma_kernel<true, true, false>
      <<<dim3(M / 64, C / 128), 256, 0, stream>>>(
          oh, wprojh, bproj, d_out, M, C, C);
}
